// CondAttLSTM_56676388438187
// MI455X (gfx1250) — matrix operand order repaired and checked by reading, not compiled, not run on hardware
//
#include <hip/hip_runtime.h>

// Problem constants (from reference)
#define DIN 512
#define DD  512
#define DCC 512
#define AA  256
#define BB  16
#define TT  64
#define LL  64
#define G4  2048  // 4*D

typedef __bf16 bf16;
typedef __bf16 v16bf __attribute__((ext_vector_type(16)));
typedef __bf16 v8bf  __attribute__((ext_vector_type(8)));
typedef float  v8f   __attribute__((ext_vector_type(8)));

// packed-weight pool offsets (bf16 elements); all matrices have K=512 (16 K-tiles)
#define PK_U    ((size_t)0)
#define PK_C    ((size_t)1048576)
#define PK_P    ((size_t)2097152)
#define PK_H    ((size_t)3145728)
#define PK_AH   ((size_t)4194304)   // att_h_W1    (512x256)
#define PK_HH1  ((size_t)4325376)   // hatt_h_W1   (512x256)
#define PK_HHW  ((size_t)4456448)   // hatt_hist_W (512x256)
#define PK_TOT  ((size_t)4587520)

__device__ __forceinline__ v8f wmma_bf16(v16bf a, v16bf b, v8f c) {
  // D = A(16x32) * B(32x16) + C, bf16 in / f32 accumulate
  return __builtin_amdgcn_wmma_f32_16x16x32_bf16(false, a, false, b, (short)0, c,
                                                 false, false);
}

__device__ __forceinline__ v16bf cat8(v8bf lo, v8bf hi) {
  return __builtin_shufflevector(lo, hi, 0, 1, 2, 3, 4, 5, 6, 7, 8, 9, 10, 11,
                                 12, 13, 14, 15);
}

__device__ __forceinline__ float fast_tanh(float x) {
  float e = __expf(-2.f * fabsf(x));
  float r = (1.f - e) / (1.f + e);
  return __builtin_copysignf(r, x);
}
__device__ __forceinline__ float sigm(float x) { return 1.f / (1.f + __expf(-x)); }

// ---- WMMA operand loaders ---------------------------------------------------
// A tile 16(M)x32(K), per-lane data = one contiguous run K=[kh,kh+16),
// kh = (lane>=16)*16.  Vectorized: 2 x 32B f32 / 2 x 16B bf16.
__device__ __forceinline__ v16bf load_a_f32(const float* __restrict__ src, int ld,
                                            int mbase, int kbase, int lane) {
  const float* row = src + (size_t)(mbase + (lane & 15)) * ld + kbase +
                     (lane >> 4) * 16;
  v8f lo = *(const v8f*)row;         // 32B aligned (ld, kbase multiples of 8/32)
  v8f hi = *(const v8f*)(row + 8);
  v16bf a;
#pragma unroll
  for (int j = 0; j < 8; ++j) {
    a[j] = (bf16)lo[j];
    a[j + 8] = (bf16)hi[j];
  }
  return a;
}
__device__ __forceinline__ v16bf load_a_lds(const bf16* __restrict__ src, int ld,
                                            int kbase, int lane) {
  const bf16* row = src + (size_t)(lane & 15) * ld + kbase + (lane >> 4) * 16;
  v8bf lo = *(const v8bf*)row;       // ds_load_b128
  v8bf hi = *(const v8bf*)(row + 8);
  return cat8(lo, hi);
}
// B tile 32(K)x16(N) from f32 row-major (prep-kernel one-shot GEMMs only).
__device__ __forceinline__ v16bf load_b_f32(const float* __restrict__ src, int ld,
                                            int kbase, int nbase, int lane) {
  v16bf b;
  int n = nbase + (lane & 15);
  int kb = kbase + (lane >> 4) * 16;
#pragma unroll
  for (int r = 0; r < 8; ++r) {
    b[2 * r]     = (bf16)src[(size_t)(kb + 2 * r) * ld + n];
    b[2 * r + 1] = (bf16)src[(size_t)(kb + 2 * r + 1) * ld + n];
  }
  return b;
}
// B tile from pre-packed bf16 pool: lane's 16 values are one contiguous 32B run.
// chunk id = (ntile*16 + kt)*32 + lane.
__device__ __forceinline__ v16bf load_b_pk(const bf16* __restrict__ pk, int ntile,
                                           int kt, int lane) {
  const bf16* p = pk + (((size_t)(ntile * 16 + kt)) * 32 + lane) * 16;
  v8bf lo = *(const v8bf*)p;         // global_load_b128
  v8bf hi = *(const v8bf*)(p + 8);
  return cat8(lo, hi);
}

// ---- Kernel 1: grid-parallel precompute + weight packing + hist zero --------
// blocks [0,1024):     x4 = X@Wx + bx                (WMMA)
// blocks [1024,1152):  ctx_trans = context@aW1 + ab1 (WMMA)
// blocks [1152,1280):  zero hist (ws is poisoned every call)
// blocks [1280,2400):  pack U,C,P,H,ahW1,hhW1,hhW -> bf16 WMMA-B layout
__global__ __launch_bounds__(256) void prep_kernel(
    const float* __restrict__ X, const float* __restrict__ Wx,
    const float* __restrict__ bx, const float* __restrict__ ctx,
    const float* __restrict__ aW1, const float* __restrict__ ab1,
    const float* __restrict__ U, const float* __restrict__ Cw,
    const float* __restrict__ Pw, const float* __restrict__ Hw,
    const float* __restrict__ ahW1, const float* __restrict__ hhW1,
    const float* __restrict__ hhW, float* __restrict__ x4,
    float* __restrict__ ctx_trans, float* __restrict__ hist,
    bf16* __restrict__ pkbase) {
  int blk = blockIdx.x;
  int lane = threadIdx.x & 31;
  int wv = threadIdx.x >> 5;
  if (blk < 1024) {
    int tile = blk * 8 + wv;  // 0..8191
    int mt = tile >> 7, nt = tile & 127;
    v8f acc = {0.f, 0.f, 0.f, 0.f, 0.f, 0.f, 0.f, 0.f};
    for (int kt = 0; kt < 16; ++kt) {
      v16bf a = load_a_f32(X, DIN, mt * 16, kt * 32, lane);
      v16bf b = load_b_f32(Wx, G4, kt * 32, nt * 16, lane);
      acc = wmma_bf16(a, b, acc);
    }
    int n = nt * 16 + (lane & 15);
    float bias = bx[n];
#pragma unroll
    for (int r = 0; r < 8; ++r) {
      int m = mt * 16 + (lane >> 4) * 8 + r;
      x4[(size_t)m * G4 + n] = acc[r] + bias;
    }
  } else if (blk < 1152) {
    int tile = (blk - 1024) * 8 + wv;  // 0..1023
    int mt = tile >> 4, nt = tile & 15;
    v8f acc = {0.f, 0.f, 0.f, 0.f, 0.f, 0.f, 0.f, 0.f};
    for (int kt = 0; kt < 16; ++kt) {
      v16bf a = load_a_f32(ctx, DCC, mt * 16, kt * 32, lane);
      v16bf b = load_b_f32(aW1, AA, kt * 32, nt * 16, lane);
      acc = wmma_bf16(a, b, acc);
    }
    int n = nt * 16 + (lane & 15);
    float bias = ab1[n];
#pragma unroll
    for (int r = 0; r < 8; ++r) {
      int m = mt * 16 + (lane >> 4) * 8 + r;
      ctx_trans[(size_t)m * AA + n] = acc[r] + bias;
    }
  } else if (blk < 1280) {
    int i0 = (blk - 1152) * 256 + threadIdx.x;  // stride 32768
    for (int i = i0; i < BB * TT * DD; i += 32768) hist[i] = 0.f;
  } else {
    // weight packing: one 16-element lane-chunk per thread
    int cid = (blk - 1280) * 256 + (int)threadIdx.x;  // 0..286719
    const float* W;
    bf16* dst;
    int ld, c;
    if (cid < 262144) {                 // U,C,P,H: 65536 chunks each
      int mi = cid >> 16;
      c = cid & 65535;
      W = (mi == 0) ? U : (mi == 1) ? Cw : (mi == 2) ? Pw : Hw;
      dst = pkbase + (size_t)mi * 1048576;
      ld = G4;
    } else {                            // ahW1,hhW1,hhW: 8192 chunks each
      int c2 = cid - 262144;
      int mi = c2 >> 13;
      c = c2 & 8191;
      W = (mi == 0) ? ahW1 : (mi == 1) ? hhW1 : hhW;
      dst = pkbase + PK_AH + (size_t)mi * 131072;
      ld = AA;
    }
    int l2 = c & 31;
    int kt = (c >> 5) & 15;
    int nt = c >> 9;
    int n = nt * 16 + (l2 & 15);
    int kb = kt * 32 + (l2 >> 4) * 16;
    bf16* d = dst + (size_t)c * 16;
#pragma unroll
    for (int j = 0; j < 16; ++j) d[j] = (bf16)W[(size_t)(kb + j) * ld + n];
  }
}

// ---- Kernel 2: persistent single-workgroup sequential scan ------------------
__device__ __forceinline__ void gate_pass(const bf16* __restrict__ A_lds,
                                          const bf16* __restrict__ Bpk,
                                          v8f acc[4], int wv, int lane) {
  for (int kt = 0; kt < 16; ++kt) {
    v16bf a = load_a_lds(A_lds, DD, kt * 32, lane);
    if (kt < 15)
      __builtin_prefetch(Bpk + (((size_t)(wv * 16 + kt + 1)) * 32) * 16, 0, 1);
#pragma unroll
    for (int g = 0; g < 4; ++g) {
      v16bf b = load_b_pk(Bpk, g * 32 + wv, kt, lane);
      acc[g] = wmma_bf16(a, b, acc[g]);
    }
  }
}

__global__ __launch_bounds__(1024, 1) void scan_kernel(
    const float* __restrict__ x4, const float* __restrict__ ctx_trans,
    float* __restrict__ hist, float* __restrict__ hist_att,
    const bf16* __restrict__ pkbase, const float* __restrict__ ctx,
    const int* __restrict__ parent, const int* __restrict__ cmask,
    const float* __restrict__ maskp, const float* __restrict__ aW2,
    const float* __restrict__ ab2p, const float* __restrict__ hhb,
    const float* __restrict__ hW2, const float* __restrict__ hb2p,
    float* __restrict__ hs, float* __restrict__ cs, float* __restrict__ ctxs) {
  __shared__ bf16 s_hbf[BB * DD];   // h_t (WMMA A source)            16 KB
  __shared__ bf16 s_abuf[BB * DD];  // staged A: ctx_vec/par_h/h_ctx  16 KB
  __shared__ bf16 s_q[2 * BB * AA]; // hq | (hhq + hatt_hist_b)       16 KB
  __shared__ float s_catt[BB * LL]; // ctx attention raw -> weights    4 KB
  __shared__ float s_hw[BB * LL];   // hist attention raw -> weights   4 KB

  const int tid = threadIdx.x, lane = tid & 31, wv = tid >> 5;
  const int bb = tid >> 6, ll = tid & 63;
  const float ab2 = ab2p[0], hb2 = hb2p[0];
  const bf16* Upk = pkbase + PK_U;
  const bf16* Cpk = pkbase + PK_C;
  const bf16* Ppk = pkbase + PK_P;
  const bf16* Hpk = pkbase + PK_H;
  const bf16* AHpk = pkbase + PK_AH;
  const bf16* HH1pk = pkbase + PK_HH1;
  const bf16* HHWpk = pkbase + PK_HHW;

  float h_reg[8], c_reg[8];
#pragma unroll
  for (int r = 0; r < 8; ++r) h_reg[r] = c_reg[r] = 0.f;
  for (int i = tid; i < BB * DD; i += 1024) s_hbf[i] = (bf16)0.f;
  __syncthreads();

  for (int t = 0; t < TT; ++t) {
    // ---- phase 1: hq = h@att_h_W1 ; hhq = h@hatt_h_W1 + hatt_hist_b (WMMA)
    {
      int tile = wv & 15;
      const bf16* Bpk = (wv < 16) ? AHpk : HH1pk;
      v8f acc = {0.f, 0.f, 0.f, 0.f, 0.f, 0.f, 0.f, 0.f};
      for (int kt = 0; kt < 16; ++kt) {
        v16bf a = load_a_lds(s_hbf, DD, kt * 32, lane);
        v16bf b = load_b_pk(Bpk, tile, kt, lane);
        acc = wmma_bf16(a, b, acc);
      }
      int n = tile * 16 + (lane & 15);
      bf16* dst = (wv < 16) ? s_q : (s_q + BB * AA);
      float bias = (wv < 16) ? 0.f : hhb[n];
#pragma unroll
      for (int r = 0; r < 8; ++r) {
        int m = (lane >> 4) * 8 + r;
        dst[m * AA + n] = (bf16)(acc[r] + bias);
      }
    }
    __syncthreads();
    // ---- phase 2: attention logits (one thread per (b,l))
    {
      float rw;
      if (cmask[bb * LL + ll] > 0) {
        rw = -3.0e38f;
      } else {
        rw = ab2;
        const float* ct = ctx_trans + (size_t)(bb * LL + ll) * AA;
        const bf16* q0 = s_q + bb * AA;
        for (int a = 0; a < AA; ++a)
          rw += fast_tanh(ct[a] + (float)q0[a]) * aW2[a];
      }
      s_catt[bb * LL + ll] = rw;
      if (ll < t) {
        float hr = hb2;
        const float* ha = hist_att + (size_t)(bb * TT + ll) * AA;
        const bf16* q1 = s_q + BB * AA + bb * AA;
        for (int a = 0; a < AA; ++a)
          hr += fast_tanh(ha[a] + (float)q1[a]) * hW2[a];
        s_hw[bb * LL + ll] = hr;
      }
    }
    __syncthreads();
    // ---- phase 3: softmaxes (serial per-b, 16+16 threads)
    if (tid < 16) {
      int b = tid;
      float mx = -3.0e38f;
      for (int l = 0; l < LL; ++l) mx = fmaxf(mx, s_catt[b * LL + l]);
      float s = 0.f;
      for (int l = 0; l < LL; ++l) {
        float v = s_catt[b * LL + l];
        float e = (v > -1.0e37f) ? __expf(v - mx) : 0.f;
        s_catt[b * LL + l] = e;
        s += e;
      }
      float inv = 1.f / s;
      for (int l = 0; l < LL; ++l) s_catt[b * LL + l] *= inv;
    } else if (tid < 32) {
      int b = tid - 16;
      if (t > 0) {
        float mx = -3.0e38f;
        for (int l = 0; l < t; ++l) mx = fmaxf(mx, s_hw[b * LL + l]);
        float s = 0.f;
        for (int l = 0; l < t; ++l) {
          float e = __expf(s_hw[b * LL + l] - mx);
          s_hw[b * LL + l] = e;
          s += e;
        }
        float inv = 1.f / (s + 1e-7f);
        for (int l = 0; l < t; ++l) s_hw[b * LL + l] *= inv;
      }
    }
    __syncthreads();

    // ---- gate accumulators: init from x4[:, t, :]
    v8f acc[4];
#pragma unroll
    for (int g = 0; g < 4; ++g) {
#pragma unroll
      for (int r = 0; r < 8; ++r) {
        int m = (lane >> 4) * 8 + r;
        acc[g][r] = x4[(size_t)(m * TT + t) * G4 + g * DD + wv * 16 + (lane & 15)];
      }
    }
    // pass 0: h @ U
    gate_pass(s_hbf, Upk, acc, wv, lane);
    // fill ctx_vec -> abuf (+ ctxs output), pass 1: ctx_vec @ C
    {
#pragma unroll
      for (int i = 0; i < 8; ++i) {
        int d = ll + 64 * i;
        float cv = 0.f;
        for (int l = 0; l < LL; ++l)
          cv += s_catt[bb * LL + l] * ctx[(size_t)(bb * LL + l) * DCC + d];
        ctxs[(size_t)(bb * TT + t) * DCC + d] = cv;
        s_abuf[bb * DD + d] = (bf16)cv;
      }
    }
    __syncthreads();
    gate_pass(s_abuf, Cpk, acc, wv, lane);
    __syncthreads();
    // fill par_h -> abuf, pass 2: par_h @ P
    {
      int pr = parent[bb * TT + t];
#pragma unroll
      for (int i = 0; i < 8; ++i) {
        int d = ll + 64 * i;
        float ph = (t > 0) ? hist[(size_t)(bb * TT + pr) * DD + d] : 0.f;
        s_abuf[bb * DD + d] = (bf16)ph;
      }
    }
    __syncthreads();
    gate_pass(s_abuf, Ppk, acc, wv, lane);
    __syncthreads();
    // fill h_ctx -> abuf, pass 3: h_ctx @ H
    {
#pragma unroll
      for (int i = 0; i < 8; ++i) {
        int d = ll + 64 * i;
        float hc = 0.f;
        for (int l = 0; l < t; ++l)
          hc += s_hw[bb * LL + l] * hist[(size_t)(bb * TT + l) * DD + d];
        s_abuf[bb * DD + d] = (bf16)hc;
      }
    }
    __syncthreads();
    gate_pass(s_abuf, Hpk, acc, wv, lane);
    __syncthreads();

    // ---- gate nonlinearity + state update (WMMA D layout, no LDS bounce)
    {
      int col = wv * 16 + (lane & 15);
#pragma unroll
      for (int r = 0; r < 8; ++r) {
        int m = (lane >> 4) * 8 + r;  // batch index
        float ii = sigm(acc[0][r]);
        float ff = sigm(acc[1][r]);
        float oo = sigm(acc[3][r]);
        float cn = ff * c_reg[r] + ii * fast_tanh(acc[2][r]);
        float hn = oo * fast_tanh(cn);
        float mm = maskp[m * TT + t];
        hn = (1.f - mm) * h_reg[r] + mm * hn;
        cn = (1.f - mm) * c_reg[r] + mm * cn;
        h_reg[r] = hn;
        c_reg[r] = cn;
        s_hbf[m * DD + col] = (bf16)hn;
        size_t o = (size_t)(m * TT + t) * DD + col;
        hist[o] = hn;
        hs[o] = hn;
        cs[o] = cn;
      }
    }
    __syncthreads();
    // ---- append hist_att[:, t, :] = h_new @ hatt_hist_W (incremental cache)
    if (wv < 16) {
      v8f a2 = {0.f, 0.f, 0.f, 0.f, 0.f, 0.f, 0.f, 0.f};
      for (int kt = 0; kt < 16; ++kt) {
        v16bf a = load_a_lds(s_hbf, DD, kt * 32, lane);
        v16bf b = load_b_pk(HHWpk, wv, kt, lane);
        a2 = wmma_bf16(a, b, a2);
      }
      int n = wv * 16 + (lane & 15);
#pragma unroll
      for (int r = 0; r < 8; ++r) {
        int m = (lane >> 4) * 8 + r;
        hist_att[(size_t)(m * TT + t) * AA + n] = a2[r];
      }
    }
    __syncthreads();
  }
}

extern "C" void kernel_launch(void* const* d_in, const int* in_sizes, int n_in,
                              void* d_out, int out_size, void* d_ws, size_t ws_size,
                              hipStream_t stream) {
  (void)in_sizes; (void)n_in; (void)out_size; (void)ws_size;
  const float* X    = (const float*)d_in[0];
  const float* ctx  = (const float*)d_in[1];
  const int* parent = (const int*)d_in[2];
  const int* cmask  = (const int*)d_in[3];
  const float* mask = (const float*)d_in[4];
  const float* Wx   = (const float*)d_in[5];
  const float* bx   = (const float*)d_in[6];
  const float* U    = (const float*)d_in[7];
  const float* Cw   = (const float*)d_in[8];
  const float* Pw   = (const float*)d_in[9];
  const float* Hw   = (const float*)d_in[10];
  const float* aW1  = (const float*)d_in[11];
  const float* ab1  = (const float*)d_in[12];
  const float* ahW1 = (const float*)d_in[13];
  const float* aW2  = (const float*)d_in[14];
  const float* ab2  = (const float*)d_in[15];
  const float* hhW  = (const float*)d_in[16];
  const float* hhb  = (const float*)d_in[17];
  const float* hhW1 = (const float*)d_in[18];
  const float* hW2  = (const float*)d_in[19];
  const float* hb2  = (const float*)d_in[20];

  // workspace layout: x4 | ctx_trans | hist | hist_att (f32) | packed bf16 pool
  float* ws        = (float*)d_ws;
  float* x4        = ws;
  float* ctx_trans = x4 + (size_t)BB * TT * G4;
  float* hist      = ctx_trans + (size_t)BB * LL * AA;
  float* hist_att  = hist + (size_t)BB * TT * DD;
  bf16* pkbase     = (bf16*)(hist_att + (size_t)BB * TT * AA);

  float* hs   = (float*)d_out;
  float* cs   = hs + (size_t)BB * TT * DD;
  float* ctxs = cs + (size_t)BB * TT * DD;

  prep_kernel<<<2400, 256, 0, stream>>>(X, Wx, bx, ctx, aW1, ab1, U, Cw, Pw, Hw,
                                        ahW1, hhW1, hhW, x4, ctx_trans, hist,
                                        pkbase);
  scan_kernel<<<1, 1024, 0, stream>>>(x4, ctx_trans, hist, hist_att, pkbase, ctx,
                                      parent, cmask, mask, aW2, ab2, hhb, hW2,
                                      hb2, hs, cs, ctxs);
}
